// LSTM_3418793968171
// MI455X (gfx1250) — compile-verified, matrix operand order checked by reading, not run on hardware
//
#include <hip/hip_runtime.h>

// ---------------------------------------------------------------------------
// Fused persistent-kernel LSTM for MI455X (gfx1250, wave32, WMMA bf16).
//
//   B=32, T=2048, E=H=512.  out = [h(32x512) ; c(32x512)] f32.
//
//   - x packed once to bf16 WMMA A-tile layout (67 MB; read once ~3us @23.3TB/s).
//   - W = concat(Wx,Wh) [1024 x 2048] packed to bf16 B-tile layout (4 MB).
//     Each of 32 WGs owns 16 hidden columns x 4 gates and keeps its 128 KB
//     weight slice resident in LDS (of 320 KB/WGP) for all 2048 steps.
//   - h ping-pongs through L2 as a 32 KB pre-swizzled bf16 buffer; c stays in
//     LDS. Device-wide step sync = monotonic atomic counter + agent fences.
//   - Dual WMMA accumulators break the RAW hazard chain (bf16 WMMA->WMMA
//     needs ~5 slots, ISA 7.12.1); next x-tile prefetched during the barrier.
// ---------------------------------------------------------------------------

typedef __attribute__((ext_vector_type(16))) __bf16 v16bf;
typedef __attribute__((ext_vector_type(8)))  float  v8f;

#define EMB      512
#define HID      512
#define BATCH    32
#define TLEN     2048
#define NWG      32
#define NTHREADS 256

// workspace layout (bytes)
#define XBF_OFF   ((size_t)0)
#define XBF_BYTES ((size_t)BATCH * TLEN * EMB * 2)                 // 67,108,864
#define WBF_OFF   (XBF_OFF + XBF_BYTES)
#define WBF_BYTES ((size_t)128 * 32 * 512 * 2)                     //  4,194,304
#define HP_OFF    (WBF_OFF + WBF_BYTES)
#define HP_BYTES  ((size_t)2 * 2 * 16 * 32 * 16 * 2)               //     65,536
#define CTR_OFF   (HP_OFF + HP_BYTES)

// LDS carve (dynamic): B tiles | G tiles | c state
#define LDS_B_BYTES (4 * 32 * 512 * 2)    // 131072: [gate][kc][lane][p] bf16
#define LDS_G_BYTES (2 * 4 * 16 * 16 * 4) //   8192: [mt][gate][m][n] f32
#define LDS_C_BYTES (2 * 16 * 16 * 4)     //   2048: [mt][m][n] f32
#define LDS_BYTES   (LDS_B_BYTES + LDS_G_BYTES + LDS_C_BYTES)

__device__ __forceinline__ unsigned short f2bf(float f) {
  union { float f; unsigned u; } v; v.f = f;
  unsigned r = v.u + 0x7fffu + ((v.u >> 16) & 1u);   // RNE
  return (unsigned short)(r >> 16);
}

// fast gate activations on the serial critical path (v_exp_f32 + v_rcp_f32)
__device__ __forceinline__ float fast_sigmoid(float x) {
  return __frcp_rn(1.0f + __expf(-x));
}
__device__ __forceinline__ float fast_tanh(float x) {
  // tanh(x) = 2*sigmoid(2x) - 1
  return __builtin_fmaf(2.0f, __frcp_rn(1.0f + __expf(-2.0f * x)), -1.0f);
}

// ---------------------------------------------------------------------------
// Pack x[f32,(b,t,e)] -> bf16 A-tile layout: [t][mt(2)][kc(16)][lane(32)][p(16)]
// 16-bit A 16x32 layout (ISA 7.12.2): grp=(k%32)/8, lane=m+16*(grp&1),
// p=(grp>>1)*8 + k%8.
// ---------------------------------------------------------------------------
__global__ void pack_x_kernel(const float* __restrict__ x,
                              unsigned short* __restrict__ xbf) {
  size_t d = (size_t)blockIdx.x * NTHREADS + threadIdx.x;
  int p    = (int)(d & 15);
  int lane = (int)((d >> 4) & 31);
  int kc   = (int)((d >> 9) & 15);
  int mt   = (int)((d >> 13) & 1);
  size_t t = d >> 14;
  int grp = ((p >> 3) << 1) | (lane >> 4);
  int k   = kc * 32 + grp * 8 + (p & 7);
  int b   = mt * 16 + (lane & 15);
  xbf[d] = f2bf(x[((size_t)b * TLEN + t) * EMB + k]);
}

// ---------------------------------------------------------------------------
// Pack W = concat(Wx,Wh) -> bf16 B-tile layout: [nt(128)][kc(32)][lane(32)][p(16)]
// (B 32x16 mirrors A with N on lanes.)
// ---------------------------------------------------------------------------
__global__ void pack_w_kernel(const float* __restrict__ Wx,
                              const float* __restrict__ Wh,
                              unsigned short* __restrict__ wbf) {
  size_t d = (size_t)blockIdx.x * NTHREADS + threadIdx.x;
  int p    = (int)(d & 15);
  int lane = (int)((d >> 4) & 31);
  int kc   = (int)((d >> 9) & 31);
  int nt   = (int)(d >> 14);
  int grp = ((p >> 3) << 1) | (lane >> 4);
  int k   = kc * 32 + grp * 8 + (p & 7);
  int n   = nt * 16 + (lane & 15);
  float v = (k < EMB) ? Wx[(size_t)k * (4 * HID) + n]
                      : Wh[(size_t)(k - EMB) * (4 * HID) + n];
  wbf[d] = f2bf(v);
}

__global__ void init_kernel(unsigned short* __restrict__ hpack,
                            unsigned* __restrict__ ctr) {
  int i = blockIdx.x * NTHREADS + threadIdx.x;
  if (i < 2 * 16384) hpack[i] = 0;   // h0 = 0 in both ping-pong buffers
  if (i == 0) *ctr = 0;
}

// ---------------------------------------------------------------------------
// Persistent recurrence kernel: 32 WGs x 256 threads (8 waves).
// wave w: gate g = w&3, M-tile mt = w>>2. WG bid owns hidden cols [bid*16,+16).
// ---------------------------------------------------------------------------
__launch_bounds__(NTHREADS)
__global__ void lstm_kernel(const unsigned short* __restrict__ xbf,
                            const unsigned short* __restrict__ wbf,
                            unsigned short* __restrict__ hpack,
                            const float* __restrict__ bx,
                            const float* __restrict__ bh,
                            float* __restrict__ out,
                            unsigned* __restrict__ ctr) {
  extern __shared__ char smem[];
  unsigned short* Bl = (unsigned short*)smem;                       // [4][32][32][16]
  float* G = (float*)(smem + LDS_B_BYTES);                          // [2][4][16][16]
  float* C = (float*)(smem + LDS_B_BYTES + LDS_G_BYTES);            // [2][16][16]

  const int tid  = threadIdx.x;
  const int bid  = blockIdx.x;
  const int lane = tid & 31;
  const int w    = tid >> 5;
  const int g    = w & 3;
  const int mt   = w >> 2;

  // ---- prologue: pull this WG's 128 KB weight slice into LDS (resident)
  for (int gg = 0; gg < 4; ++gg) {
    const float4* src = (const float4*)(const void*)(wbf + (size_t)(gg * 32 + bid) * (32 * 512));
    float4* dst = (float4*)(void*)(Bl + gg * (32 * 512));
    for (int i = tid; i < (32 * 512 * 2) / 16; i += NTHREADS) dst[i] = src[i];
  }
  for (int i = tid; i < 512; i += NTHREADS) C[i] = 0.0f;   // c0 = 0
  __syncthreads();

  // per-lane gate-column bias (time invariant)
  const int n0 = lane & 15;
  const int m0 = (lane >> 4) * 8;
  const int colg = g * HID + bid * 16 + n0;
  const float bias = bx[colg] + bh[colg];

  unsigned short* const hbuf0 = hpack;
  unsigned short* const hbuf1 = hpack + 16384;

  union V16 { v16bf v; float4 q[2]; };

  for (int t = 0; t < TLEN; ++t) {
    const unsigned short* hsrc = (t & 1) ? hbuf1 : hbuf0;
    unsigned short*       hdst = (t & 1) ? hbuf0 : hbuf1;

    const unsigned short* xA = xbf + ((size_t)t * 2 + mt) * (16 * 512) + lane * 16;
    const unsigned short* hA = hsrc + (size_t)mt * (16 * 512) + lane * 16;
    const unsigned short* Bw = Bl + g * (32 * 512) + lane * 16;

    // two independent accumulators -> two interleavable WMMA streams
    v8f acc0 = {};
    v8f acc1 = {};

    // K = 0..511: x_t @ Wx slice
#pragma unroll 2
    for (int kc = 0; kc < 16; kc += 2) {
      v16bf a0 = *(const v16bf*)(const void*)(xA + kc * 512);
      v16bf a1 = *(const v16bf*)(const void*)(xA + (kc + 1) * 512);
      V16 b0; const float4* bq0 = (const float4*)(const void*)(Bw + kc * 512);
      b0.q[0] = bq0[0]; b0.q[1] = bq0[1];
      V16 b1; const float4* bq1 = (const float4*)(const void*)(Bw + (kc + 1) * 512);
      b1.q[0] = bq1[0]; b1.q[1] = bq1[1];
      acc0 = __builtin_amdgcn_wmma_f32_16x16x32_bf16(false, a0, false, b0.v,
                                                     (short)0, acc0, false, false);
      acc1 = __builtin_amdgcn_wmma_f32_16x16x32_bf16(false, a1, false, b1.v,
                                                     (short)0, acc1, false, false);
    }
    // K = 512..1023: h_{t-1} @ Wh slice
#pragma unroll 2
    for (int kc = 0; kc < 16; kc += 2) {
      v16bf a0 = *(const v16bf*)(const void*)(hA + kc * 512);
      v16bf a1 = *(const v16bf*)(const void*)(hA + (kc + 1) * 512);
      V16 b0; const float4* bq0 = (const float4*)(const void*)(Bw + (kc + 16) * 512);
      b0.q[0] = bq0[0]; b0.q[1] = bq0[1];
      V16 b1; const float4* bq1 = (const float4*)(const void*)(Bw + (kc + 17) * 512);
      b1.q[0] = bq1[0]; b1.q[1] = bq1[1];
      acc0 = __builtin_amdgcn_wmma_f32_16x16x32_bf16(false, a0, false, b0.v,
                                                     (short)0, acc0, false, false);
      acc1 = __builtin_amdgcn_wmma_f32_16x16x32_bf16(false, a1, false, b1.v,
                                                     (short)0, acc1, false, false);
    }

    // prefetch next step's x-tile (overlaps gate math + barrier spin)
    if (t + 1 < TLEN)
      __builtin_prefetch(xA + 2 * 16 * 512, 0, 1);

    // combine accumulators, scatter D tile (+bias) to LDS: element (m0+r, n0)
    float* Gp = G + (mt * 4 + g) * 256;
#pragma unroll
    for (int r = 0; r < 8; ++r) Gp[(m0 + r) * 16 + n0] = (acc0[r] + acc1[r]) + bias;
    __syncthreads();

    // gate math: 512 cells over 256 threads (serial critical path -> fast ops)
    for (int idx = tid; idx < 512; idx += NTHREADS) {
      int emt = idx >> 8, m = (idx >> 4) & 15, n = idx & 15;
      const float* Gm = G + emt * 1024;
      float gf = Gm[0 * 256 + m * 16 + n];
      float gi = Gm[1 * 256 + m * 16 + n];
      float gc = Gm[2 * 256 + m * 16 + n];
      float go = Gm[3 * 256 + m * 16 + n];
      float f  = fast_sigmoid(gf);
      float i  = fast_sigmoid(gi);
      float cc = fast_tanh(gc);
      float o  = fast_sigmoid(go);
      float c  = __builtin_fmaf(f, C[idx], i * cc);
      C[idx] = c;
      float h = o * fast_tanh(c);

      // repack h into bf16 A-tile layout for next step's WMMA A operand
      int col = bid * 16 + n;
      int kc  = col >> 5, kk = col & 31, grp = kk >> 3;
      int l2  = m + ((grp & 1) << 4);
      int pp  = ((grp >> 1) << 3) + (kk & 7);
      hdst[((emt * 16 + kc) * 32 + l2) * 16 + pp] = f2bf(h);

      if (t == TLEN - 1) {
        int b = emt * 16 + m;
        out[(size_t)b * HID + col] = h;                              // h_final
        out[(size_t)BATCH * HID + (size_t)b * HID + col] = c;        // c_final
      }
    }

    // ---- device-wide step barrier (monotonic counter; release/acquire) ----
    __threadfence();          // release: h stores agent-visible
    __syncthreads();
    if (tid == 0) {
      atomicAdd(ctr, 1u);
      const unsigned target = (unsigned)(t + 1) * NWG;
      while (__hip_atomic_load(ctr, __ATOMIC_RELAXED, __HIP_MEMORY_SCOPE_AGENT) < target)
        __builtin_amdgcn_s_sleep(2);
      __threadfence();        // acquire: invalidate so new h is observed
    }
    __syncthreads();
  }
}

// ---------------------------------------------------------------------------
extern "C" void kernel_launch(void* const* d_in, const int* in_sizes, int n_in,
                              void* d_out, int out_size, void* d_ws, size_t ws_size,
                              hipStream_t stream) {
  const float* x  = (const float*)d_in[0];   // [32,2048,512]
  const float* Wx = (const float*)d_in[1];   // [512,2048]
  const float* Wh = (const float*)d_in[2];   // [512,2048]
  const float* bx = (const float*)d_in[3];   // [2048]
  const float* bh = (const float*)d_in[4];   // [2048]
  float* out = (float*)d_out;                // [h(32x512); c(32x512)]

  char* ws = (char*)d_ws;
  unsigned short* xbf   = (unsigned short*)(ws + XBF_OFF);
  unsigned short* wbf   = (unsigned short*)(ws + WBF_OFF);
  unsigned short* hpack = (unsigned short*)(ws + HP_OFF);
  unsigned*       ctr   = (unsigned*)(ws + CTR_OFF);

  pack_x_kernel<<<dim3((BATCH * TLEN * EMB) / NTHREADS), dim3(NTHREADS), 0, stream>>>(x, xbf);
  pack_w_kernel<<<dim3((128 * 32 * 512) / NTHREADS), dim3(NTHREADS), 0, stream>>>(Wx, Wh, wbf);
  init_kernel<<<dim3(128), dim3(NTHREADS), 0, stream>>>(hpack, ctr);
  lstm_kernel<<<dim3(NWG), dim3(NTHREADS), LDS_BYTES, stream>>>(
      xbf, wbf, hpack, bx, bh, out, ctr);
}
